// ScaledDotProductAttention_60936995996031
// MI455X (gfx1250) — matrix operand order repaired and checked by reading, not compiled, not run on hardware
//
#include <hip/hip_runtime.h>

typedef _Float16 v16h __attribute__((ext_vector_type(16)));
typedef _Float16 v8h  __attribute__((ext_vector_type(8)));
typedef float    v8f  __attribute__((ext_vector_type(8)));
typedef float    v4f  __attribute__((ext_vector_type(4)));

#define B_ 4
#define H_ 16
#define L_ 2048
#define D_ 64
#define KB 64
// softmax runs in the exp2 domain: fold scale * log2(e) into the score
#define SCALE_L2E (0.125f * 1.4426950408889634f)
#define NEG_BIAS  (-1.44e9f)   /* NEG_INF * log2(e) */

__device__ __forceinline__ v16h join16(v8h a, v8h b) {
  return __builtin_shufflevector(a, b, 0,1,2,3,4,5,6,7,8,9,10,11,12,13,14,15);
}

__device__ __forceinline__ v8f zero8() {
  v8f z;
  #pragma unroll
  for (int i = 0; i < 8; ++i) z[i] = 0.0f;
  return z;
}

__device__ __forceinline__ v8f splat8(float x) {
  v8f z;
  #pragma unroll
  for (int i = 0; i < 8; ++i) z[i] = x;
  return z;
}

// lane[i] <- lane[i^k] within each 16-lane row, on the VALU (v_permlane16_b32):
// no DS pipe, no DScnt waits.
__device__ __forceinline__ float pl16(float v, unsigned s0, unsigned s1) {
  unsigned u = __builtin_bit_cast(unsigned, v);
  u = __builtin_amdgcn_permlane16(u, u, s0, s1, false, false);
  return __builtin_bit_cast(float, u);
}

// nibble selectors for xor-1/2/4/8 butterflies within a 16-lane row
__constant__ constexpr unsigned kSel[4][2] = {
  {0x67452301u, 0xEFCDAB89u},   // i ^ 1
  {0x54761032u, 0xDCFE98BAu},   // i ^ 2
  {0x32107654u, 0xBA98FEDCu},   // i ^ 4
  {0xFEDCBA98u, 0x76543210u},   // i ^ 8
};

__global__ __launch_bounds__(256)
void fattn_f16_wmma(const float* __restrict__ Q,
                    const float* __restrict__ K,
                    const float* __restrict__ V,
                    const int*   __restrict__ M,
                    float* __restrict__ O) {
  // double-buffered K/V tiles: stage tile i+1 while computing tile i
  __shared__ _Float16 Kt[2][64][72];     // [buf][key][d]  (row pad: 144B = 9x16B)
  __shared__ _Float16 Vt[2][64][72];     // [buf][d][key]  (transposed)
  __shared__ _Float16 Ps[8][16][72];     // per-wave P scratch [qrow][key]
  __shared__ float    Bias[2][64];

  const int tid  = threadIdx.x;
  const int lane = tid & 31;
  const int wave = tid >> 5;
  const int r    = lane & 15;
  const int hi   = lane >> 4;

  const int bh = blockIdx.y;             // b*H + h
  const int b  = bh >> 4;                // H_ == 16
  const int q0 = blockIdx.x * 128 + wave * 16;

  const float* Qh = Q + (size_t)bh * L_ * D_;
  const float* Kh = K + (size_t)bh * L_ * D_;
  const float* Vh = V + (size_t)bh * L_ * D_;
  const int*   Mb = M + (size_t)b * L_;

  const int srow = tid >> 2;             // staging: key row 0..63
  const int scol = (tid & 3) * 16;       // staging: d col 0,16,32,48

  // stage one K/V tile (f32 -> f16; V transposed) into buffer `buf`
  auto stage = [&](int kb, int buf) {
    const float* ks = Kh + (size_t)(kb + srow) * D_ + scol;
    const float* vs = Vh + (size_t)(kb + srow) * D_ + scol;
    v4f k0 = *(const v4f*)ks,       k1 = *(const v4f*)(ks + 4);
    v4f k2 = *(const v4f*)(ks + 8), k3 = *(const v4f*)(ks + 12);
    v4f a0 = *(const v4f*)vs,       a1 = *(const v4f*)(vs + 4);
    v4f a2 = *(const v4f*)(vs + 8), a3 = *(const v4f*)(vs + 12);
    v8h h0, h1;
    #pragma unroll
    for (int i = 0; i < 4; ++i) {
      h0[i] = (_Float16)k0[i]; h0[4 + i] = (_Float16)k1[i];
      h1[i] = (_Float16)k2[i]; h1[4 + i] = (_Float16)k3[i];
    }
    *(v8h*)&Kt[buf][srow][scol]     = h0;
    *(v8h*)&Kt[buf][srow][scol + 8] = h1;
    #pragma unroll
    for (int i = 0; i < 4; ++i) {
      Vt[buf][scol + i][srow]      = (_Float16)a0[i];
      Vt[buf][scol + 4 + i][srow]  = (_Float16)a1[i];
      Vt[buf][scol + 8 + i][srow]  = (_Float16)a2[i];
      Vt[buf][scol + 12 + i][srow] = (_Float16)a3[i];
    }
    if (tid < KB) Bias[buf][tid] = Mb[kb + tid] ? 0.0f : NEG_BIAS;
    if (kb + 2 * KB < L_) {            // warm L1 for the tile staged two rounds later
      __builtin_prefetch(Kh + (size_t)(kb + 2 * KB + srow) * D_ + scol, 0, 1);
      __builtin_prefetch(Vh + (size_t)(kb + 2 * KB + srow) * D_ + scol, 0, 1);
    }
  };

  // ---- Q A-fragments: 16 rows x 64 d -> two K=32 chunks, f32->f16 ----
  v16h qf[2];
  {
    const float* qrow = Qh + (size_t)(q0 + r) * D_;
    #pragma unroll
    for (int c = 0; c < 2; ++c) {
      const float* p0 = qrow + c * 32 + hi * 8;        // K = 8*hi .. +8
      const float* p1 = qrow + c * 32 + 16 + hi * 8;   // K = 16+8*hi .. +8
      v8h a, bv;
      #pragma unroll
      for (int i = 0; i < 8; ++i) { a[i] = (_Float16)p0[i]; bv[i] = (_Float16)p1[i]; }
      qf[c] = join16(a, bv);
    }
  }

  // all-ones B fragment: row sums of P via the matrix pipe
  v16h ones;
  #pragma unroll
  for (int i = 0; i < 16; ++i) ones[i] = (_Float16)1.0f;

  v8f Oa[4];
  #pragma unroll
  for (int d = 0; d < 4; ++d) Oa[d] = zero8();
  v8f mrow = splat8(-3.0e38f);
  v8f lrow = zero8();

  stage(0, 0);                            // prologue: tile 0 into buffer 0

  for (int kb = 0; kb < L_; kb += KB) {
    const int buf = (kb >> 6) & 1;
    __syncthreads();                      // stage(kb) done; buf^1 free for reuse
    if (kb + KB < L_) stage(kb + KB, buf ^ 1);   // overlaps with compute below

    // ---- preload ALL K fragments, then run the QK WMMA chain ----
    v16h kf[4][2];
    #pragma unroll
    for (int t = 0; t < 4; ++t) {
      #pragma unroll
      for (int c = 0; c < 2; ++c) {
        const _Float16* kp = &Kt[buf][t * 16 + r][c * 32 + hi * 8];
        kf[t][c] = join16(*(const v8h*)kp, *(const v8h*)(kp + 16));
      }
    }
    v8f S[4];
    #pragma unroll
    for (int t = 0; t < 4; ++t) {
      v8f acc = zero8();
      acc  = __builtin_amdgcn_wmma_f32_16x16x32_f16(false, qf[0], false, kf[t][0],
                                                    (short)0, acc, false, false);
      S[t] = __builtin_amdgcn_wmma_f32_16x16x32_f16(false, qf[1], false, kf[t][1],
                                                    (short)0, acc, false, false);
    }

    // scores in exp2 domain: s = S*scale*log2e + bias2 (vector form)
    v8f sv[4];
    #pragma unroll
    for (int t = 0; t < 4; ++t)
      sv[t] = S[t] * SCALE_L2E + Bias[buf][t * 16 + r];
    v8f mxv = __builtin_elementwise_max(
                  __builtin_elementwise_max(sv[0], sv[1]),
                  __builtin_elementwise_max(sv[2], sv[3]));

    // 16-lane butterfly max on the VALU (v_permlane16_b32)
    #pragma unroll
    for (int s = 0; s < 4; ++s) {
      v8f tv;
      #pragma unroll
      for (int j = 0; j < 8; ++j) tv[j] = pl16(mxv[j], kSel[s][0], kSel[s][1]);
      mxv = __builtin_elementwise_max(mxv, tv);
    }

    const v8f mnew = __builtin_elementwise_max(mrow, mxv);
    v8f alphav;
    #pragma unroll
    for (int j = 0; j < 8; ++j) alphav[j] = __builtin_amdgcn_exp2f(mrow[j] - mnew[j]);
    mrow = mnew;

    // P = exp2(s - m), stored f16 to per-wave scratch (C layout -> LDS)
    #pragma unroll
    for (int t = 0; t < 4; ++t) {
      const v8f d8 = sv[t] - mnew;
      #pragma unroll
      for (int j = 0; j < 8; ++j) {
        const float p = __builtin_amdgcn_exp2f(d8[j]);
        Ps[wave][j + 8 * hi][t * 16 + r] = (_Float16)p;
      }
    }

    // rescale running output by alpha (overlaps with P-store latency)
    #pragma unroll
    for (int d = 0; d < 4; ++d) Oa[d] = Oa[d] * alphav;

    // per-wave LDS round trip: C-layout -> A-layout needs cross-lane data
    asm volatile("s_wait_dscnt 0" ::: "memory");

    // P A-fragments (two K=32 chunks) + preload all V fragments
    v16h pf[2];
    #pragma unroll
    for (int c = 0; c < 2; ++c) {
      const _Float16* pp = &Ps[wave][r][c * 32];
      pf[c] = join16(*(const v8h*)(pp + 8 * hi), *(const v8h*)(pp + 16 + 8 * hi));
    }
    v16h vf[4][2];
    #pragma unroll
    for (int d = 0; d < 4; ++d) {
      #pragma unroll
      for (int c = 0; c < 2; ++c) {
        const _Float16* vp = &Vt[buf][d * 16 + r][c * 32 + hi * 8];
        vf[d][c] = join16(*(const v8h*)vp, *(const v8h*)(vp + 16));
      }
    }

    // ---- O += P * V ----
    #pragma unroll
    for (int d = 0; d < 4; ++d) {
      Oa[d] = __builtin_amdgcn_wmma_f32_16x16x32_f16(false, pf[0], false, vf[d][0],
                                                     (short)0, Oa[d], false, false);
      Oa[d] = __builtin_amdgcn_wmma_f32_16x16x32_f16(false, pf[1], false, vf[d][1],
                                                     (short)0, Oa[d], false, false);
    }

    // ---- row sums of P via WMMA with all-ones B: no cross-lane shuffles ----
    v8f rs = zero8();
    rs = __builtin_amdgcn_wmma_f32_16x16x32_f16(false, pf[0], false, ones,
                                                (short)0, rs, false, false);
    rs = __builtin_amdgcn_wmma_f32_16x16x32_f16(false, pf[1], false, ones,
                                                (short)0, rs, false, false);
    lrow = lrow * alphav + rs;
  }

  // ---- normalize and store (C layout: row = j + 8*hi, col = d*16 + r) ----
  #pragma unroll
  for (int j = 0; j < 8; ++j) {
    const float inv = 1.0f / lrow[j];
    float* orow = O + ((size_t)bh * L_ + q0 + j + 8 * hi) * D_ + r;
    #pragma unroll
    for (int d = 0; d < 4; ++d) orow[d * 16] = Oa[d][j] * inv;
  }
}

extern "C" void kernel_launch(void* const* d_in, const int* in_sizes, int n_in,
                              void* d_out, int out_size, void* d_ws, size_t ws_size,
                              hipStream_t stream) {
  const float* Q = (const float*)d_in[0];
  const float* K = (const float*)d_in[1];
  const float* V = (const float*)d_in[2];
  const int*   M = (const int*)d_in[3];
  float* Outp = (float*)d_out;
  dim3 grid(L_ / 128, B_ * H_);   // 16 query-blocks x 64 (b,h) heads
  fattn_f16_wmma<<<grid, 256, 0, stream>>>(Q, K, V, M, Outp);
}
